// GNN_Model_43885975831068
// MI455X (gfx1250) — matrix operand order repaired and checked by reading, not compiled, not run on hardware
//
#include <hip/hip_runtime.h>

#define NP    32768
#define NFE   131072
#define NEDGE 2000000
#define EH    1000000
#define GCN   128
#define KPAD  256

typedef _Float16 half16 __attribute__((ext_vector_type(16)));
typedef _Float16 half8  __attribute__((ext_vector_type(8)));
typedef float    f32x8  __attribute__((ext_vector_type(8)));

// ---------------- decoder GEMMs ----------------

// out[b][n] = x_p[b][:] . w_pd[n][:] + b_pd[n]; reshape(-1,3) -> node = b*4096 + n/3, ch = n%3
__global__ void gemm_pd_kernel(const float* __restrict__ xP, const float* __restrict__ W,
                               const float* __restrict__ bias, float* __restrict__ xpOut) {
  const int n = blockIdx.x;            // 0..12287
  const int tid = threadIdx.x;
  float acc[8] = {0.f,0.f,0.f,0.f,0.f,0.f,0.f,0.f};
  const float* wrow = W + (long)n * 12288;
  for (int k = tid * 4; k < 12288; k += 1024) {
    float4 wv = *(const float4*)(wrow + k);
#pragma unroll
    for (int b = 0; b < 8; ++b) {
      float4 xv = *(const float4*)(xP + b * 12288 + k);
      acc[b] += wv.x*xv.x + wv.y*xv.y + wv.z*xv.z + wv.w*xv.w;
    }
  }
  __shared__ float red[256];
#pragma unroll
  for (int b = 0; b < 8; ++b) {
    red[tid] = acc[b];
    __syncthreads();
    for (int s = 128; s > 0; s >>= 1) {
      if (tid < s) red[tid] += red[tid + s];
      __syncthreads();
    }
    if (tid == 0) {
      int node = b * 4096 + n / 3;
      int ch = n % 3;
      xpOut[(long)node * GCN + ch] = red[0] + bias[n];
    }
    __syncthreads();
  }
}

// out (2048,192) -> node = r*64 + n/3, ch = n%3
__global__ void gemm_fd_kernel(const float* __restrict__ xF, const float* __restrict__ W,
                               const float* __restrict__ bias, float* __restrict__ xfOut) {
  int gid = blockIdx.x * 256 + threadIdx.x;
  if (gid >= 2048 * 192) return;
  int r = gid / 192, n = gid % 192;
  float acc = bias[n];
  const float* xr = xF + r * 64;
  const float* wr = W + n * 64;
#pragma unroll 8
  for (int k = 0; k < 64; ++k) acc += xr[k] * wr[k];
  int node = r * 64 + n / 3, ch = n % 3;
  xfOut[(long)node * GCN + ch] = acc;
}

// ---------------- CSR build (once per launch; reused by all 3 layers) ----------------

// degP over ei_f dst (row1, even cols); degF over ei_b dst (row0, odd cols)
__global__ void edge_count(const int* __restrict__ ei, int* __restrict__ degP,
                           int* __restrict__ degF) {
  int e = blockIdx.x * 256 + threadIdx.x;
  if (e >= EH) return;
  atomicAdd(degP + ei[NEDGE + 2 * e], 1);
  atomicAdd(degF + ei[2 * e + 1], 1);
}

// exclusive prefix sum, single workgroup (n multiple of 256)
__global__ void exscan_kernel(const int* __restrict__ deg, int* __restrict__ rowStart, int n) {
  __shared__ int sums[256];
  const int tid = threadIdx.x;
  const int chunk = n / 256;
  const int base = tid * chunk;
  int s = 0;
  for (int i = 0; i < chunk; ++i) s += deg[base + i];
  sums[tid] = s;
  __syncthreads();
  if (tid == 0) {
    int acc = 0;
    for (int i = 0; i < 256; ++i) { int t = sums[i]; sums[i] = acc; acc += t; }
  }
  __syncthreads();
  int run = sums[tid];
  for (int i = 0; i < chunk; ++i) { rowStart[base + i] = run; run += deg[base + i]; }
}

__global__ void copy_int(const int* __restrict__ a, int* __restrict__ b, int n) {
  int i = blockIdx.x * 256 + threadIdx.x;
  if (i < n) b[i] = a[i];
}

// bucket edges into CSR adjacency (2M atomics total, once per launch)
__global__ void csr_fill(const int* __restrict__ ei, int* __restrict__ curP,
                         int* __restrict__ srcP, int* __restrict__ curF,
                         int* __restrict__ srcF) {
  int e = blockIdx.x * 256 + threadIdx.x;
  if (e >= EH) return;
  int sF = ei[2 * e],             dP = ei[NEDGE + 2 * e];       // forward edge
  int sP = ei[NEDGE + 2 * e + 1], dF = ei[2 * e + 1];           // backward edge
  int p = atomicAdd(curP + dP, 1); srcP[p] = sF;
  int q = atomicAdd(curF + dF, 1); srcF[q] = sP;
}

// ---------------- gather-based aggregation (no atomics) ----------------

// one wave per node, 128-dim source features; writes raw sum
__global__ void gather128(const float* __restrict__ xsrc, const int* __restrict__ rowStart,
                          const int* __restrict__ deg, const int* __restrict__ srcList,
                          float* __restrict__ agg, long N) {
  long gid = (long)blockIdx.x * 256 + threadIdx.x;
  long node = gid >> 5;
  int lane = (int)(gid & 31);
  if (node >= N) return;
  int start = rowStart[node];
  int d = deg[node];
  float4 acc = make_float4(0.f, 0.f, 0.f, 0.f);
  for (int i = 0; i < d; ++i) {
    int s = srcList[start + i];
    float4 v = *(const float4*)(xsrc + (long)s * GCN + lane * 4);
    acc.x += v.x; acc.y += v.y; acc.z += v.z; acc.w += v.w;
  }
  *(float4*)(agg + node * GCN + lane * 4) = acc;
}

// one thread per node, 3-dim source features (layer-0 forward only)
__global__ void gather3(const float* __restrict__ xsrc, const int* __restrict__ rowStart,
                        const int* __restrict__ deg, const int* __restrict__ srcList,
                        float* __restrict__ agg, long N) {
  long node = (long)blockIdx.x * 256 + threadIdx.x;
  if (node >= N) return;
  int start = rowStart[node];
  int d = deg[node];
  float a0 = 0.f, a1 = 0.f, a2 = 0.f;
  for (int i = 0; i < d; ++i) {
    const float* sp = xsrc + (long)srcList[start + i] * GCN;
    a0 += sp[0]; a1 += sp[1]; a2 += sp[2];
  }
  float* dp = agg + node * GCN;
  dp[0] = a0; dp[1] = a1; dp[2] = a2;
}

// ---------------- staging to f16 (K padded to 256) ----------------

// staged[n][k] = k<dimA ? agg[n][k]/max(deg,1) : (k<dimA+dimX ? xdst[n][k-dimA] : 0)
__global__ void stage_cat(const float* __restrict__ agg, const int* __restrict__ deg,
                          const float* __restrict__ xdst, _Float16* __restrict__ out,
                          int dimA, int dimX, long N) {
  long gid = (long)blockIdx.x * 256 + threadIdx.x;
  long nrow = gid >> 8;
  int k = (int)(gid & 255);
  if (nrow >= N) return;
  float v = 0.0f;
  if (k < dimA) {
    int d = deg[nrow];
    float c = d > 1 ? (float)d : 1.0f;
    v = agg[nrow * GCN + k] / c;
  } else if (k < dimA + dimX) {
    v = xdst[nrow * GCN + (k - dimA)];
  }
  out[gid] = (_Float16)v;
}

// Wc[o][k] = k<dimA ? wl[o][k] : (k<dimA+dimX ? wr[o][k-dimA] : 0)
__global__ void wcat_kernel(const float* __restrict__ wl, const float* __restrict__ wr,
                            _Float16* __restrict__ Wc, int dimA, int dimX) {
  int gid = blockIdx.x * 256 + threadIdx.x;  // 128*256
  int o = gid >> 8;
  int k = gid & 255;
  float v = 0.0f;
  if (k < dimA)             v = wl[o * dimA + k];
  else if (k < dimA + dimX) v = wr[o * dimX + (k - dimA)];
  Wc[gid] = (_Float16)v;
}

// ---------------- WMMA dense: Y = relu(staged @ Wc.T + bias) ----------------

__global__ void sage_dense_wmma(const _Float16* __restrict__ A,  // N x 256 f16
                                const _Float16* __restrict__ W,  // 128 x 256 f16
                                const float* __restrict__ bias,  // 128
                                float* __restrict__ Y,           // N x 128
                                int N) {
  const int lane = threadIdx.x & 31;
  const int wave = threadIdx.x >> 5;
  const int tile = blockIdx.x * 8 + wave;
  const int row0 = (tile >> 3) * 16;
  const int col0 = (tile & 7) * 16;
  if (row0 >= N) return;  // wave-uniform; EXEC stays all-ones for WMMA
  const int hi = lane >> 4;
  const int m = lane & 15;
  const long arow = (long)(row0 + m) * KPAD;
  const long brow = (long)(col0 + m) * KPAD;  // B column n = W row n
  f32x8 c = {};
#pragma unroll
  for (int k0 = 0; k0 < KPAD; k0 += 32) {
    half8 alo = *(const half8*)(A + arow + k0 + hi * 8);
    half8 ahi = *(const half8*)(A + arow + k0 + 16 + hi * 8);
    half8 blo = *(const half8*)(W + brow + k0 + hi * 8);
    half8 bhi = *(const half8*)(W + brow + k0 + 16 + hi * 8);
    half16 av, bv;
#pragma unroll
    for (int i = 0; i < 8; ++i) {
      av[i] = alo[i]; av[i + 8] = ahi[i];
      bv[i] = blo[i]; bv[i + 8] = bhi[i];
    }
    c = __builtin_amdgcn_wmma_f32_16x16x32_f16(false, av, false, bv, (short)0, c,
                                               false, false);
  }
  const int col = col0 + m;
  const float bb = bias[col];
#pragma unroll
  for (int v = 0; v < 8; ++v) {
    int row = row0 + hi * 8 + v;   // C: VGPR v -> M = v + 8*hi
    float val = c[v] + bb;
    Y[(long)row * GCN + col] = val > 0.0f ? val : 0.0f;
  }
}

// ---------------- LayerNorm (one wave per 128-wide row) ----------------

__global__ void ln_kernel(float* __restrict__ x, const float* __restrict__ g,
                          const float* __restrict__ bta, long N) {
  int lane = threadIdx.x & 31;
  int wave = threadIdx.x >> 5;
  long row = (long)blockIdx.x * 8 + wave;
  if (row >= N) return;
  float4 v = *(float4*)(x + row * GCN + lane * 4);
  float s = v.x + v.y + v.z + v.w;
#pragma unroll
  for (int m = 16; m > 0; m >>= 1) s += __shfl_xor(s, m, 32);
  float mean = s * (1.0f / 128.0f);
  float dx = v.x - mean, dy = v.y - mean, dz = v.z - mean, dw = v.w - mean;
  float sq = dx * dx + dy * dy + dz * dz + dw * dw;
#pragma unroll
  for (int m = 16; m > 0; m >>= 1) sq += __shfl_xor(sq, m, 32);
  float inv = rsqrtf(sq * (1.0f / 128.0f) + 1e-5f);
  int c = lane * 4;
  float4 o;
  o.x = dx * inv * g[c + 0] + bta[c + 0];
  o.y = dy * inv * g[c + 1] + bta[c + 1];
  o.z = dz * inv * g[c + 2] + bta[c + 2];
  o.w = dw * inv * g[c + 3] + bta[c + 3];
  *(float4*)(x + row * GCN + c) = o;
}

// ---------------- conv1d as WMMA GEMM ----------------

__global__ void cvt_f16(const float* __restrict__ a, _Float16* __restrict__ b, int n) {
  int i = blockIdx.x * 256 + threadIdx.x;
  if (i < n) b[i] = (_Float16)a[i];
}

// out[b][o][t] = relu(bias[o] + sum_{c,k} x[b][c][t+k] * w[o][c*8+k])
// GEMM: M = t (16-row tiles), N = o (64 -> 4 tiles), K = C*8
__global__ void conv_wmma(const float* __restrict__ x, const _Float16* __restrict__ wh,
                          const float* __restrict__ bias, float* __restrict__ y,
                          int C, int L, int Lout, int TR) {
  const int lane = threadIdx.x & 31;
  const int wave = threadIdx.x >> 5;
  const int tile = blockIdx.x * 8 + wave;
  const int total = 8 * TR * 4;
  if (tile >= total) return;           // wave-uniform
  const int b  = tile / (TR * 4);
  const int rt = (tile >> 2) % TR;
  const int ct = tile & 3;
  const int row0 = rt * 16;
  const int col0 = ct * 16;
  const int hi = lane >> 4;
  const int m = lane & 15;
  const int t = row0 + m;              // A row (time index)
  const int o = col0 + m;              // B column (output channel)
  const int Kc = C * 8;
  const float* xb = x + (long)b * C * L;
  const _Float16* wrow = wh + (long)o * Kc;
  f32x8 c = {};
  for (int k0 = 0; k0 < Kc; k0 += 32) {
    // A octets: K = k0+hi*8+j and k0+16+hi*8+j  ->  c = K>>3, value x[b][c][t + (K&7)]
    const int cA0 = (k0 + hi * 8) >> 3;
    const int cA1 = (k0 + 16 + hi * 8) >> 3;
    const float* xa0 = xb + (long)cA0 * L + t;
    const float* xa1 = xb + (long)cA1 * L + t;
    half16 av;
#pragma unroll
    for (int j = 0; j < 8; ++j) {
      av[j]     = (_Float16)xa0[j];
      av[j + 8] = (_Float16)xa1[j];
    }
    // B: lane holds column o, K octets k0+hi*8+j and k0+16+hi*8+j (same as A)
    half8 blo = *(const half8*)(wrow + k0 + hi * 8);
    half8 bhi = *(const half8*)(wrow + k0 + 16 + hi * 8);
    half16 bv;
#pragma unroll
    for (int j = 0; j < 8; ++j) { bv[j] = blo[j]; bv[j + 8] = bhi[j]; }
    c = __builtin_amdgcn_wmma_f32_16x16x32_f16(false, av, false, bv, (short)0, c,
                                               false, false);
  }
  const float bb = bias[o];
  float* yo = y + ((long)b * 64 + o) * Lout;
#pragma unroll
  for (int v = 0; v < 8; ++v) {
    int tr = row0 + hi * 8 + v;
    if (tr < Lout) {                   // mask padding rows of the last tile
      float val = c[v] + bb;
      yo[tr] = val > 0.0f ? val : 0.0f;
    }
  }
}

// ---------------- dense head ----------------

__global__ void d1_kernel(const float* __restrict__ h, const float* __restrict__ W,
                          const float* __restrict__ bias, float* __restrict__ out) {
  const int j = blockIdx.x;
  const int tid = threadIdx.x;
  float acc[8] = {0.f,0.f,0.f,0.f,0.f,0.f,0.f,0.f};
  const float4* wrow = (const float4*)(W + (long)j * 260800);
  for (int kk = tid; kk < 65200; kk += 256) {
    float4 wv = wrow[kk];
#pragma unroll
    for (int b = 0; b < 8; ++b) {
      float4 xv = *(const float4*)(h + (long)b * 260800 + kk * 4);
      acc[b] += wv.x*xv.x + wv.y*xv.y + wv.z*xv.z + wv.w*xv.w;
    }
  }
  __shared__ float red[256];
#pragma unroll
  for (int b = 0; b < 8; ++b) {
    red[tid] = acc[b];
    __syncthreads();
    for (int s = 128; s > 0; s >>= 1) {
      if (tid < s) red[tid] += red[tid + s];
      __syncthreads();
    }
    if (tid == 0) {
      float v = red[0] + bias[j];
      out[b * 100 + j] = v > 0.0f ? v : 0.0f;
    }
    __syncthreads();
  }
}

__global__ void d2_softmax(const float* __restrict__ h, const float* __restrict__ W,
                           const float* __restrict__ bias, float* __restrict__ out) {
  int b = threadIdx.x;
  if (b >= 8) return;
  float z0 = bias[0], z1 = bias[1];
  for (int j = 0; j < 100; ++j) {
    float v = h[b * 100 + j];
    z0 += v * W[j];
    z1 += v * W[100 + j];
  }
  float mx = fmaxf(z0, z1);
  float e0 = __expf(z0 - mx), e1 = __expf(z1 - mx);
  float s = e0 + e1;
  out[b * 2 + 0] = e0 / s;
  out[b * 2 + 1] = e1 / s;
}

// ---------------- launch ----------------

extern "C" void kernel_launch(void* const* d_in, const int* in_sizes, int n_in,
                              void* d_out, int out_size, void* d_ws, size_t ws_size,
                              hipStream_t stream) {
  (void)in_sizes; (void)n_in; (void)out_size; (void)ws_size;
  const float* x_p  = (const float*)d_in[0];
  const float* x_f  = (const float*)d_in[1];
  const int*   ei   = (const int*)d_in[2];
  const float* w_pd = (const float*)d_in[3];
  const float* b_pd = (const float*)d_in[4];
  const float* w_fd = (const float*)d_in[5];
  const float* b_fd = (const float*)d_in[6];
  const float* wl1[3] = {(const float*)d_in[7],  (const float*)d_in[10], (const float*)d_in[13]};
  const float* bl1[3] = {(const float*)d_in[8],  (const float*)d_in[11], (const float*)d_in[14]};
  const float* wr1[3] = {(const float*)d_in[9],  (const float*)d_in[12], (const float*)d_in[15]};
  const float* wl2[3] = {(const float*)d_in[16], (const float*)d_in[19], (const float*)d_in[22]};
  const float* bl2[3] = {(const float*)d_in[17], (const float*)d_in[20], (const float*)d_in[23]};
  const float* wr2[3] = {(const float*)d_in[18], (const float*)d_in[21], (const float*)d_in[24]};
  const float* ln_g = (const float*)d_in[25];
  const float* ln_b = (const float*)d_in[26];
  const float* cw1 = (const float*)d_in[27]; const float* cb1 = (const float*)d_in[28];
  const float* cw2 = (const float*)d_in[29]; const float* cb2 = (const float*)d_in[30];
  const float* cw3 = (const float*)d_in[31]; const float* cb3 = (const float*)d_in[32];
  const float* d1w = (const float*)d_in[33]; const float* d1b = (const float*)d_in[34];
  const float* d2w = (const float*)d_in[35]; const float* d2b = (const float*)d_in[36];

  char* ws = (char*)d_ws;
  size_t off = 0;
  float* xp = (float*)(ws + off);           off += (size_t)NP  * GCN * 4;   // 16 MB
  float* xf = (float*)(ws + off);           off += (size_t)NFE * GCN * 4;   // 64 MB
  float* agg = (float*)(ws + off);          off += (size_t)NFE * GCN * 4;   // 64 MB (P/F union)
  _Float16* staged = (_Float16*)(ws + off); off += (size_t)NFE * KPAD * 2;  // 64 MB (P/F union)
  _Float16* Wc = (_Float16*)(ws + off);     off += (size_t)GCN * KPAD * 2;
  _Float16* wh = (_Float16*)(ws + off);     off += (size_t)64 * GCN * 8 * 2; // conv w f16
  int* degP = (int*)(ws + off);             off += (size_t)NP * 4;
  int* degF = (int*)(ws + off);             off += (size_t)NFE * 4;
  int* rowP = (int*)(ws + off);             off += (size_t)NP * 4;
  int* rowF = (int*)(ws + off);             off += (size_t)NFE * 4;
  int* curP = (int*)(ws + off);             off += (size_t)NP * 4;
  int* curF = (int*)(ws + off);             off += (size_t)NFE * 4;
  int* srcP = (int*)(ws + off);             off += (size_t)EH * 4;          // 4 MB
  int* srcF = (int*)(ws + off);             off += (size_t)EH * 4;          // 4 MB
  // conv intermediates alias agg; d1 output aliases staged (GNN phase finished)
  float* c1 = agg;
  float* c2 = c1 + (size_t)8 * 64 * 4089;
  float* c3 = c2 + (size_t)8 * 64 * 4082;
  float* d1o = (float*)staged;

  // ---- CSR build (once; reused by all layers) ----
  hipMemsetAsync(degP, 0, (size_t)NP * 4, stream);
  hipMemsetAsync(degF, 0, (size_t)NFE * 4, stream);
  edge_count<<<(EH + 255) / 256, 256, 0, stream>>>(ei, degP, degF);
  exscan_kernel<<<1, 256, 0, stream>>>(degP, rowP, NP);
  exscan_kernel<<<1, 256, 0, stream>>>(degF, rowF, NFE);
  copy_int<<<NP / 256, 256, 0, stream>>>(rowP, curP, NP);
  copy_int<<<NFE / 256, 256, 0, stream>>>(rowF, curF, NFE);
  csr_fill<<<(EH + 255) / 256, 256, 0, stream>>>(ei, curP, srcP, curF, srcF);

  // ---- decoders ----
  gemm_pd_kernel<<<12288, 256, 0, stream>>>(x_p, w_pd, b_pd, xp);
  gemm_fd_kernel<<<(2048 * 192 + 255) / 256, 256, 0, stream>>>(x_f, w_fd, b_fd, xf);

  // ---- 3 SAGE layers ----
  for (int i = 0; i < 3; ++i) {
    const int dimF = (i == 0) ? 3 : GCN;  // current xf feature dim
    const int dimP = (i == 0) ? 3 : GCN;  // current xp feature dim

    // forward SAGE: xf -> xp (gather, no atomics)
    if (i == 0)
      gather3<<<NP / 256, 256, 0, stream>>>(xf, rowP, degP, srcP, agg, (long)NP);
    else
      gather128<<<(NP * 32) / 256, 256, 0, stream>>>(xf, rowP, degP, srcP, agg, (long)NP);
    wcat_kernel<<<128, 256, 0, stream>>>(wl1[i], wr1[i], Wc, dimF, dimP);
    stage_cat<<<NP, 256, 0, stream>>>(agg, degP, xp, staged, dimF, dimP, (long)NP);
    sage_dense_wmma<<<(NP / 16) * 8 / 8, 256, 0, stream>>>(staged, Wc, bl1[i], xp, NP);
    if (i < 2) ln_kernel<<<NP / 8, 256, 0, stream>>>(xp, ln_g + i * GCN, ln_b + i * GCN, (long)NP);

    // backward SAGE: xp (now 128-d) -> xf
    gather128<<<(NFE * 32) / 256, 256, 0, stream>>>(xp, rowF, degF, srcF, agg, (long)NFE);
    wcat_kernel<<<128, 256, 0, stream>>>(wl2[i], wr2[i], Wc, GCN, dimF);
    stage_cat<<<NFE, 256, 0, stream>>>(agg, degF, xf, staged, GCN, dimF, (long)NFE);
    sage_dense_wmma<<<(NFE / 16) * 8 / 8, 256, 0, stream>>>(staged, Wc, bl2[i], xf, NFE);
    if (i < 2) ln_kernel<<<NFE / 8, 256, 0, stream>>>(xf, ln_g + i * GCN, ln_b + i * GCN, (long)NFE);
  }

  // ---- conv head (WMMA GEMM): xp flat is exactly (8,128,4096) ----
  // conv1: C=128, L=4096, Lout=4089, TR=256  (8*256*4 = 8192 tiles)
  cvt_f16<<<(64 * 128 * 8) / 256, 256, 0, stream>>>(cw1, wh, 64 * 128 * 8);
  conv_wmma<<<8192 / 8, 256, 0, stream>>>(xp, wh, cb1, c1, 128, 4096, 4089, 256);
  // conv2: C=64, L=4089, Lout=4082, TR=256
  cvt_f16<<<(64 * 64 * 8) / 256, 256, 0, stream>>>(cw2, wh, 64 * 64 * 8);
  conv_wmma<<<8192 / 8, 256, 0, stream>>>(c1, wh, cb2, c2, 64, 4089, 4082, 256);
  // conv3: C=64, L=4082, Lout=4075, TR=255  (8*255*4 = 8160 tiles)
  cvt_f16<<<(64 * 64 * 8) / 256, 256, 0, stream>>>(cw3, wh, 64 * 64 * 8);
  conv_wmma<<<8160 / 8, 256, 0, stream>>>(c2, wh, cb3, c3, 64, 4082, 4075, 255);

  // ---- dense head ----
  d1_kernel<<<100, 256, 0, stream>>>(c3, d1w, d1b, d1o);
  d2_softmax<<<1, 32, 0, stream>>>(d1o, d2w, d2b, (float*)d_out);
}
